// attention_46797963657749
// MI455X (gfx1250) — matrix operand order repaired and checked by reading, not compiled, not run on hardware
//
#include <hip/hip_runtime.h>
#include <math.h>

typedef __attribute__((ext_vector_type(16))) __bf16 v16bf;
typedef __attribute__((ext_vector_type(8)))  float  v8f;

#define B_    32
#define TX_   4096
#define ENC_  512
#define NS_   128
#define HID_  512
#define MT_   64            // rows per workgroup in the GEMM kernel

// round-to-nearest-even f32 -> bf16 bits
static __device__ __forceinline__ unsigned short f2bf_bits(float f) {
    union { float f; unsigned u; } v; v.f = f;
    unsigned r = v.u + 0x7FFFu + ((v.u >> 16) & 1u);
    return (unsigned short)(r >> 16);
}

// tanh via hardware transcendentals: 1 - 2/(exp(2x)+1)
static __device__ __forceinline__ float fast_tanh(float x) {
    float e = __expf(2.0f * x);
    return 1.0f - 2.0f * __builtin_amdgcn_rcpf(e + 1.0f);
}

// sigmoid via hardware transcendentals
static __device__ __forceinline__ float fast_sigmoid(float x) {
    return __builtin_amdgcn_rcpf(1.0f + __expf(-x));
}

// ---------------------------------------------------------------------------
// Repack W1 (E=512 x H=512, f32, row-major [e][h]) into bf16 WMMA B-fragments.
// Fragment layout per (ntile, ktile): 32 lanes x 16 bf16 (32B per lane).
//   lane l<16 : n = ntile*16+l,      holds K = kt*32 + 0..7, 16..23
//   lane l>=16: n = ntile*16+(l-16), holds K = kt*32 + 8..15, 24..31
// ---------------------------------------------------------------------------
__global__ __launch_bounds__(256) void pack_w1(const float* __restrict__ W1,
                                               unsigned short* __restrict__ W1b) {
    int idx = blockIdx.x * 256 + threadIdx.x;   // over 512*512
    int e = idx >> 9;          // K index
    int h = idx & 511;         // N index
    int ntile = h >> 4;
    int l     = h & 15;
    int kt    = e >> 5;
    int kk    = e & 31;
    int lane  = ((kk >> 3) & 1) ? (l + 16) : l;
    int j     = ((kk >> 4) << 3) + (kk & 7);
    int dst = ((ntile * 16 + kt) * 32 + lane) * 16 + j;
    W1b[dst] = f2bf_bits(W1[idx]);
}

// ---------------------------------------------------------------------------
// x2[b,h] = relu( concat(h_state,c_state)[b,:] @ W2[:,h] + b2[h] )
// ---------------------------------------------------------------------------
__global__ __launch_bounds__(256) void hc_proj(const float* __restrict__ h_state,
                                               const float* __restrict__ c_state,
                                               const float* __restrict__ W2,
                                               const float* __restrict__ b2,
                                               float* __restrict__ x2) {
    __shared__ float hc[2 * NS_];
    int b = blockIdx.x, tid = threadIdx.x;
    if (tid < NS_)      hc[tid] = h_state[b * NS_ + tid];
    else                hc[tid] = c_state[b * NS_ + (tid - NS_)];
    __syncthreads();
    for (int h = tid; h < HID_; h += 256) {
        float acc = b2[h];
        for (int e = 0; e < 2 * NS_; ++e)
            acc = fmaf(hc[e], W2[e * HID_ + h], acc);
        x2[b * HID_ + h] = fmaxf(acc, 0.0f);
    }
}

// ---------------------------------------------------------------------------
// Main fused kernel: per (batch, 64-row tile of T):
//   x1 = relu(enc_tile @ W1 + b1)  via v_wmma_f32_16x16x32_bf16
//   s  = sigmoid( tanh(x1 + x2) @ W3 + b3 )  (fused epilogue)
// 8 waves x 4 n-tiles = 512 channels; each wave owns 4 m-subtiles so every
// W1 B-fragment load feeds 4 WMMAs (4x less L2 traffic than M=16 tiling).
// Per wave: 16 kt x 4 n x 4 m = 256 WMMAs.
// ---------------------------------------------------------------------------
__global__ __launch_bounds__(256) void attn_scores(
    const float* __restrict__ enc, const unsigned short* __restrict__ W1b,
    const float* __restrict__ x2, const float* __restrict__ b1,
    const float* __restrict__ W3, const float* __restrict__ b3,
    float* __restrict__ scores)
{
    // A tile: [mt(4)][kt(16)][lane(32)][16] bf16 = 64 KB
    __shared__ alignas(64) unsigned short ldsA[4 * 16 * 32 * 16];
    __shared__ float s_lds[MT_];

    const int b    = blockIdx.y;
    const int t0   = blockIdx.x * MT_;
    const int tid  = threadIdx.x;
    const int lane = tid & 31;
    const int wave = tid >> 5;

    // --- Stage A tile (64 rows x 512 K) into LDS, bf16, ISA A-fragment layout:
    //   lane m   (<16): K%32 in 0..7  -> pos 0..7 ; 16..23 -> pos 8..15
    //   lane m+16     : K%32 in 8..15 -> pos 0..7 ; 24..31 -> pos 8..15
    const float* encRow = enc + ((size_t)b * TX_ + t0) * ENC_;
#pragma unroll 4
    for (int it = 0; it < 32; ++it) {
        int idx = it * 256 + tid;          // over 64 rows x 128 float4 groups
        int m   = idx >> 7;                // 0..63
        int kg  = idx & 127;
        int k   = kg * 4;
        float4 v = *reinterpret_cast<const float4*>(&encRow[(size_t)m * ENC_ + k]);
        int kt = k >> 5, kk = k & 31;
        int region = kk >> 3;              // 0..3
        int alane  = (m & 15) + ((region & 1) << 4);
        int pos    = ((region >> 1) << 3) + (kk & 7);
        int mt     = m >> 4;
        unsigned short* dst =
            &ldsA[(((mt * 16 + kt) * 32) + alane) * 16 + pos];
        uint2 packed;
        packed.x = (unsigned)f2bf_bits(v.x) | ((unsigned)f2bf_bits(v.y) << 16);
        packed.y = (unsigned)f2bf_bits(v.z) | ((unsigned)f2bf_bits(v.w) << 16);
        *reinterpret_cast<uint2*>(dst) = packed;
    }
    if (tid < MT_) s_lds[tid] = 0.0f;
    __syncthreads();

    // --- K loop: per kt load 4 A frags (LDS) + 4 B frags (L2), 16 WMMAs
    v8f acc[4][4] = {};
    for (int kt = 0; kt < 16; ++kt) {
        v16bf a[4];
#pragma unroll
        for (int mi = 0; mi < 4; ++mi)
            a[mi] = *reinterpret_cast<const v16bf*>(
                &ldsA[((mi * 16 + kt) * 32 + lane) * 16]);
#pragma unroll
        for (int j = 0; j < 4; ++j) {
            const int ntile = wave * 4 + j;
            const v16bf bf = *reinterpret_cast<const v16bf*>(
                &W1b[((ntile * 16 + kt) * 32 + lane) * 16]);
#pragma unroll
            for (int mi = 0; mi < 4; ++mi)
                acc[mi][j] = __builtin_amdgcn_wmma_f32_16x16x32_bf16(
                    false, a[mi], false, bf, (short)0, acc[mi][j], false, false);
        }
    }

    // --- Fused epilogue: relu(+b1), tanh(+x2), dot with W3.
    // C layout: vgpr r holds (M = r + 8*(lane>=16), N = lane&15).
    const int nl = lane & 15;
    float x2v[4], b1v[4], w3v[4];
#pragma unroll
    for (int j = 0; j < 4; ++j) {
        const int n = (wave * 4 + j) * 16 + nl;
        x2v[j] = x2[b * HID_ + n];
        b1v[j] = b1[n];
        w3v[j] = W3[n];
    }
#pragma unroll
    for (int mi = 0; mi < 4; ++mi) {
        float rowsum[8];
#pragma unroll
        for (int r = 0; r < 8; ++r) rowsum[r] = 0.0f;
#pragma unroll
        for (int j = 0; j < 4; ++j) {
#pragma unroll
            for (int r = 0; r < 8; ++r) {
                float x1 = fmaxf(acc[mi][j][r] + b1v[j], 0.0f);
                rowsum[r] += fast_tanh(x1 + x2v[j]) * w3v[j];
            }
        }
        // reduce across the 16 lanes that share each row
#pragma unroll
        for (int r = 0; r < 8; ++r) {
            float v = rowsum[r];
            v += __shfl_xor(v, 1, 32);
            v += __shfl_xor(v, 2, 32);
            v += __shfl_xor(v, 4, 32);
            v += __shfl_xor(v, 8, 32);
            rowsum[r] = v;
        }
        if ((lane & 15) == 0) {
            const int mbase = mi * 16 + ((lane >> 4) << 3);
#pragma unroll
            for (int r = 0; r < 8; ++r) atomicAdd(&s_lds[mbase + r], rowsum[r]);
        }
    }
    __syncthreads();

    if (tid < MT_) {
        float s = s_lds[tid] + b3[0];
        scores[(size_t)b * TX_ + t0 + tid] = fast_sigmoid(s);
    }
}

// ---------------------------------------------------------------------------
// In-place softmax over T=4096 per batch (scores live in d_out attn region)
// ---------------------------------------------------------------------------
__global__ __launch_bounds__(256) void softmax_inplace(float* __restrict__ attn) {
    __shared__ float red[256];
    const int b = blockIdx.x, tid = threadIdx.x;
    float* base = attn + (size_t)b * TX_;

    float s[16];
    float mx = -1e30f;
#pragma unroll
    for (int i = 0; i < 16; ++i) { s[i] = base[tid + i * 256]; mx = fmaxf(mx, s[i]); }
    red[tid] = mx; __syncthreads();
    for (int off = 128; off > 0; off >>= 1) {
        if (tid < off) red[tid] = fmaxf(red[tid], red[tid + off]);
        __syncthreads();
    }
    mx = red[0]; __syncthreads();

    float sum = 0.0f;
#pragma unroll
    for (int i = 0; i < 16; ++i) { s[i] = __expf(s[i] - mx); sum += s[i]; }
    red[tid] = sum; __syncthreads();
    for (int off = 128; off > 0; off >>= 1) {
        if (tid < off) red[tid] += red[tid + off];
        __syncthreads();
    }
    const float inv = 1.0f / red[0];
#pragma unroll
    for (int i = 0; i < 16; ++i) base[tid + i * 256] = s[i] * inv;
}

// ---------------------------------------------------------------------------
// context[b,e] = sum_t attn[b,t] * enc[b,t,e]   (second full read of enc)
// ---------------------------------------------------------------------------
__global__ __launch_bounds__(256) void context_kernel(
    const float* __restrict__ enc, const float* __restrict__ attn,
    float* __restrict__ ctx)
{
    __shared__ float a_lds[256];
    const int b = blockIdx.x, tid = threadIdx.x;
    const float* encB = enc + (size_t)b * TX_ * ENC_;
    float2 acc = make_float2(0.0f, 0.0f);

    for (int tc = 0; tc < TX_; tc += 256) {
        a_lds[tid] = attn[(size_t)b * TX_ + tc + tid];
        __syncthreads();
        for (int tt = 0; tt < 256; ++tt) {
            const float aw = a_lds[tt];
            const float2 e2 = *reinterpret_cast<const float2*>(
                encB + (size_t)(tc + tt) * ENC_ + 2 * tid);
            acc.x = fmaf(aw, e2.x, acc.x);
            acc.y = fmaf(aw, e2.y, acc.y);
        }
        __syncthreads();
    }
    ctx[b * ENC_ + 2 * tid]     = acc.x;
    ctx[b * ENC_ + 2 * tid + 1] = acc.y;
}

// ---------------------------------------------------------------------------
extern "C" void kernel_launch(void* const* d_in, const int* in_sizes, int n_in,
                              void* d_out, int out_size, void* d_ws, size_t ws_size,
                              hipStream_t stream) {
    (void)in_sizes; (void)n_in; (void)out_size; (void)ws_size;
    const float* enc     = (const float*)d_in[0];
    const float* h_state = (const float*)d_in[1];
    const float* c_state = (const float*)d_in[2];
    const float* W1      = (const float*)d_in[3];
    const float* b1      = (const float*)d_in[4];
    const float* W2      = (const float*)d_in[5];
    const float* b2      = (const float*)d_in[6];
    const float* W3      = (const float*)d_in[7];
    const float* b3      = (const float*)d_in[8];

    float* out  = (float*)d_out;
    float* attn = out;                         // (B, T, 1) region
    float* ctx  = out + (size_t)B_ * TX_;      // (B, 1, ENC) region

    unsigned short* W1b = (unsigned short*)d_ws;                       // 512 KB
    float* x2 = (float*)((char*)d_ws + (size_t)ENC_ * HID_ * 2);       // 64 KB

    pack_w1<<<(ENC_ * HID_) / 256, 256, 0, stream>>>(W1, W1b);
    hc_proj<<<B_, 256, 0, stream>>>(h_state, c_state, W2, b2, x2);

    dim3 grid(TX_ / MT_, B_);
    attn_scores<<<grid, 256, 0, stream>>>(enc, W1b, x2, b1, W3, b3, attn);

    softmax_inplace<<<B_, 256, 0, stream>>>(attn);
    context_kernel<<<B_, 256, 0, stream>>>(enc, attn, ctx);
}
